// CorrelationModule_23897198035028
// MI455X (gfx1250) — matrix operand order repaired and checked by reading, not compiled, not run on hardware
//
#include <hip/hip_runtime.h>
#include <hip/hip_bf16.h>
#include <stdint.h>

typedef __attribute__((ext_vector_type(16))) _Float16 v16h;
typedef __attribute__((ext_vector_type(8)))  float    v8f;

static constexpr int IMG_H = 96;
static constexpr int IMG_W = 128;
static constexpr int HWSZ  = IMG_H * IMG_W;   // 12288
static constexpr int NQ    = 49;              // (2*3+1)^2 displacements

__device__ __forceinline__ int imin(int a, int b) { return a < b ? a : b; }
__device__ __forceinline__ int imax(int a, int b) { return a > b ? a : b; }

// ---- CDNA5 async global->LDS copy (ASYNCcnt tracked, CDNA5 ISA) --------
__device__ __forceinline__ void async_copy_b128(void* lds_dst, const void* gsrc) {
  uint32_t laddr = (uint32_t)(uintptr_t)lds_dst;   // LDS byte address (addr[31:0])
  uint64_t gaddr = (uint64_t)(uintptr_t)gsrc;
  asm volatile("global_load_async_to_lds_b128 %0, %1, off"
               :: "v"(laddr), "v"(gaddr)
               : "memory");
}
__device__ __forceinline__ void wait_async0() {
#if __has_builtin(__builtin_amdgcn_s_wait_asynccnt)
  __builtin_amdgcn_s_wait_asynccnt(0);
#else
  asm volatile("s_wait_asynccnt 0x0" ::: "memory");
#endif
}

// ---------------------------------------------------------------- utilities
__global__ void cvt_f16_kernel(const float* __restrict__ src,
                               _Float16* __restrict__ dst, int n) {
  int i = blockIdx.x * blockDim.x + threadIdx.x;
  if (i < n) dst[i] = (_Float16)src[i];
}

__global__ void bn_prep_kernel(const float* __restrict__ s, const float* __restrict__ b,
                               const float* __restrict__ m, const float* __restrict__ v,
                               float* __restrict__ scale, float* __restrict__ shift, int n) {
  int i = blockIdx.x * blockDim.x + threadIdx.x;
  if (i < n) {
    float inv = s[i] * rsqrtf(v[i] + 1e-5f);
    scale[i] = inv;
    shift[i] = b[i] - m[i] * inv;
  }
}

// ------------------------------------------------- grid-sample + concat (f16)
__global__ void sample_concat_kernel(const float* __restrict__ f1,
                                     const float* __restrict__ f2,
                                     const float* __restrict__ coords,
                                     _Float16* __restrict__ X, float lvl_scale) {
  int idx = blockIdx.x * blockDim.x + threadIdx.x;
  if (idx >= NQ * HWSZ) return;
  int w = idx % IMG_W, h = (idx / IMG_W) % IMG_H, q = idx / HWSZ;
  int du = q / 7, dv = q % 7;
  int p = h * IMG_W + w;
  float gx = coords[p] * lvl_scale + (float)(du - 3);
  float gy = coords[HWSZ + p] * lvl_scale + (float)(dv - 3);
  float x0f = floorf(gx), y0f = floorf(gy);
  int ix = (int)x0f, iy = (int)y0f;
  float fx = gx - x0f, fy = gy - y0f;
  bool vx0 = (ix >= 0) & (ix <= IMG_W - 1);
  bool vx1 = (ix + 1 >= 0) & (ix + 1 <= IMG_W - 1);
  bool vy0 = (iy >= 0) & (iy <= IMG_H - 1);
  bool vy1 = (iy + 1 >= 0) & (iy + 1 <= IMG_H - 1);
  int cx0 = imin(imax(ix, 0), IMG_W - 1), cx1 = imin(imax(ix + 1, 0), IMG_W - 1);
  int cy0 = imin(imax(iy, 0), IMG_H - 1), cy1 = imin(imax(iy + 1, 0), IMG_H - 1);
  float w00 = (1.f - fx) * (1.f - fy) * ((vx0 && vy0) ? 1.f : 0.f);
  float w01 = fx * (1.f - fy) * ((vx1 && vy0) ? 1.f : 0.f);
  float w10 = (1.f - fx) * fy * ((vx0 && vy1) ? 1.f : 0.f);
  float w11 = fx * fy * ((vx1 && vy1) ? 1.f : 0.f);
  int i00 = cy0 * IMG_W + cx0, i01 = cy0 * IMG_W + cx1;
  int i10 = cy1 * IMG_W + cx0, i11 = cy1 * IMG_W + cx1;
  _Float16* Xq = X + (size_t)q * 64 * HWSZ;
  for (int c = 0; c < 32; ++c) {
    const float* f2c = f2 + c * HWSZ;
    float v = w00 * f2c[i00] + w01 * f2c[i01] + w10 * f2c[i10] + w11 * f2c[i11];
    Xq[(size_t)c * HWSZ + p] = (_Float16)f1[c * HWSZ + p];
    Xq[(size_t)(32 + c) * HWSZ + p] = (_Float16)v;
  }
}

// ------------------------------------- 3x3 conv as implicit GEMM using WMMA
// Block-cooperative K loop; OCx32 weight chunks double-buffered in LDS via
// async global->LDS copies. Tile map is static per wave (no divergent guards):
//   nt = wave % NT (B fragment shared across the wave's tiles)
//   mt = wave / NT + ti * (8/NT),  ti in [0, MT/(8/NT))
template <int IC, int OC, int IH, int IW, int STRIDE, int STRIP>
__global__ __launch_bounds__(256) void conv3x3_wmma(
    const _Float16* __restrict__ X, const _Float16* __restrict__ Wg,
    const float* __restrict__ scale, const float* __restrict__ shift,
    _Float16* __restrict__ Y) {
  constexpr int OH = IH / STRIDE, OW = IW / STRIDE;
  constexpr int K = IC * 9;
  constexpr int KC = K / 32;
  constexpr int COLS = STRIP * STRIDE + 2;
  constexpr int MT = OC / 16, NT = STRIP / 16;
  constexpr int MSTEP = 8 / NT;            // waves stacked along M
  constexpr int TPW = MT / MSTEP;          // tiles per wave (3/2/2/1)
  static_assert(MT % MSTEP == 0 && (8 % NT) == 0, "tile map must be exact");
  constexpr int STRIPS = OW / STRIP;
  constexpr int WCH = OC * 32;             // halves per weight chunk
  constexpr int WB128 = WCH / 8;           // 16B transfers per chunk

  __shared__ _Float16 sIn[IC * 3 * COLS];
  __shared__ _Float16 sW[2][WCH];

  int bid = blockIdx.x;
  int sx  = bid % STRIPS;
  int y   = (bid / STRIPS) % OH;
  int img = bid / (STRIPS * OH);
  int x0  = sx * STRIP;
  const _Float16* Xi = X + (size_t)img * IC * IH * IW;

  int tid = threadIdx.x;
  __builtin_prefetch(Wg, 0, 1);  // global_prefetch_b8

  // ---- stage input patch (3 rows x COLS x IC) into LDS, zero padded
  int iy0 = y * STRIDE - 1;
  int ix0 = x0 * STRIDE - 1;
  for (int e = tid; e < IC * 3 * COLS; e += 256) {
    int col = e % COLS;
    int r   = (e / COLS) % 3;
    int ic  = e / (3 * COLS);
    int yy = iy0 + r, xx = ix0 + col;
    _Float16 v = (_Float16)0.f;
    if (yy >= 0 && yy < IH && xx >= 0 && xx < IW)
      v = Xi[(size_t)ic * IH * IW + (size_t)yy * IW + xx];
    sIn[(ic * 3 + r) * COLS + col] = v;
  }
  // ---- stage weight chunk 0 synchronously
  for (int i = tid; i < WCH; i += 256)
    sW[0][i] = Wg[(size_t)(i >> 5) * K + (i & 31)];
  __syncthreads();

  int wave = tid >> 5;
  int lane = tid & 31;
  int m    = lane & 15;      // A row (outC) / D column (pixel)
  int hi   = lane >> 4;      // selects K sub-range (A) / M sub-range (D)
  int nt   = wave % NT;      // wave-static pixel tile
  int mt0  = wave / NT;      // wave-static base outC tile

  v8f acc[TPW];
#pragma unroll
  for (int ti = 0; ti < TPW; ++ti) { v8f z = {}; acc[ti] = z; }

  int buf = 0;
#pragma unroll 2
  for (int kc = 0; kc < KC; ++kc) {
    // async prefetch of next weight chunk into the other buffer
    if (kc + 1 < KC) {
      const _Float16* gW = Wg + (kc + 1) * 32;
      for (int i = tid; i < WB128; i += 256) {
        int oc = i >> 2, part = i & 3;
        async_copy_b128(&sW[buf ^ 1][oc * 32 + part * 8],
                        gW + (size_t)oc * K + part * 8);
      }
    }
    // ---- B fragment (shared by all tiles of this wave): column = lane%16, K in table order
    v16h b;
#pragma unroll
    for (int e = 0; e < 16; ++e) {
      int kk  = kc * 32 + (e < 8 ? hi * 8 + e : 16 + hi * 8 + (e - 8));
      int ic  = kk / 9;
      int tap = kk % 9;
      int ky = tap / 3, kx = tap % 3;
      b[e] = sIn[(ic * 3 + ky) * COLS + kx + (nt * 16 + m) * STRIDE];
    }

    const uint32_t* wbase = (const uint32_t*)&sW[buf][0];
#pragma unroll
    for (int ti = 0; ti < TPW; ++ti) {
      int mt = mt0 + ti * MSTEP;
      // A fragment from LDS weight chunk (two 16B reads per lane)
      union { v16h h; uint32_t u[8]; } a;
      const uint32_t* w32 = wbase + (mt * 16 + m) * 16;  // 32 halves per row
#pragma unroll
      for (int vv = 0; vv < 4; ++vv) {
        a.u[vv]     = w32[hi * 4 + vv];
        a.u[vv + 4] = w32[8 + hi * 4 + vv];
      }
      acc[ti] = __builtin_amdgcn_wmma_f32_16x16x32_f16(
          false, a.h, false, b, (short)0, acc[ti], false, false);
    }
    if (kc + 1 < KC) {
      wait_async0();       // s_wait_asynccnt 0
      __syncthreads();
      buf ^= 1;
    }
  }

  // ---- epilogue: BN + ReLU; D layout -> oc = mt*16 + r + hi*8, px = nt*16+m
  int px = x0 + nt * 16 + m;
#pragma unroll
  for (int ti = 0; ti < TPW; ++ti) {
    int mt = mt0 + ti * MSTEP;
#pragma unroll
    for (int r = 0; r < 8; ++r) {
      int oc = mt * 16 + r + hi * 8;
      float v = acc[ti][r] * scale[oc] + shift[oc];
      v = v > 0.f ? v : 0.f;
      Y[((size_t)img * OC + oc) * OH * OW + (size_t)y * OW + px] = (_Float16)v;
    }
  }
}

// ---------------------------------- ConvTranspose2d 64->32, k4 s2 p1 + BN+ReLU
__global__ void convt5_kernel(const _Float16* __restrict__ Xin,  // [49][64][48][64]
                              const _Float16* __restrict__ Wt,   // [64][32][4][4]
                              const float* __restrict__ scale,
                              const float* __restrict__ shift,
                              _Float16* __restrict__ Y) {         // [49][32][96][128]
  int idx = blockIdx.x * blockDim.x + threadIdx.x;
  if (idx >= NQ * 32 * HWSZ) return;
  int wo = idx % IMG_W, ho = (idx / IMG_W) % IMG_H;
  int oc = (idx / HWSZ) % 32, q = idx / (32 * HWSZ);
  const _Float16* Xq = Xin + (size_t)q * 64 * 48 * 64;
  float acc = 0.f;
  int ph = (ho + 1) & 1, pw = (wo + 1) & 1;
  for (int kh = ph; kh < 4; kh += 2) {
    int hi = (ho + 1 - kh) >> 1;
    if (hi < 0 || hi >= 48) continue;
    for (int kw = pw; kw < 4; kw += 2) {
      int wi = (wo + 1 - kw) >> 1;
      if (wi < 0 || wi >= 64) continue;
      const _Float16* xp = Xq + hi * 64 + wi;
      const _Float16* wp = Wt + oc * 16 + kh * 4 + kw;
#pragma unroll 8
      for (int ic = 0; ic < 64; ++ic)
        acc += (float)xp[(size_t)ic * 48 * 64] * (float)wp[(size_t)ic * 512];
    }
  }
  float v = acc * scale[oc] + shift[oc];
  Y[idx] = (_Float16)(v > 0.f ? v : 0.f);
}

// ------------------------------------------- conv6: 32->1, 3x3, pad 1 + bias
__global__ void conv6_kernel(const _Float16* __restrict__ Xin,  // [49][32][96][128]
                             const float* __restrict__ w,        // [32][3][3]
                             const float* __restrict__ bias,     // [1]
                             float* __restrict__ cost) {          // [49][96][128]
  int idx = blockIdx.x * blockDim.x + threadIdx.x;
  if (idx >= NQ * HWSZ) return;
  int x = idx % IMG_W, y = (idx / IMG_W) % IMG_H, q = idx / HWSZ;
  const _Float16* Xq = Xin + (size_t)q * 32 * HWSZ;
  float acc = bias[0];
  for (int ic = 0; ic < 32; ++ic) {
    const _Float16* xc = Xq + (size_t)ic * HWSZ;
    const float* wc = w + ic * 9;
#pragma unroll
    for (int ky = 0; ky < 3; ++ky) {
      int yy = y + ky - 1;
      if (yy < 0 || yy >= IMG_H) continue;
#pragma unroll
      for (int kx = 0; kx < 3; ++kx) {
        int xx = x + kx - 1;
        if (xx < 0 || xx >= IMG_W) continue;
        acc += (float)xc[yy * IMG_W + xx] * wc[ky * 3 + kx];
      }
    }
  }
  cost[idx] = acc;
}

// --------------------------------- DAP: out[p] = sum_q dap[p][q] * cost[q]
__global__ void dap_kernel(const float* __restrict__ cost,  // [49][12288]
                           const float* __restrict__ dw,    // [49][49]
                           float* __restrict__ out) {        // [49][12288]
  int idx = blockIdx.x * blockDim.x + threadIdx.x;
  if (idx >= NQ * HWSZ) return;
  int s = idx % HWSZ, p = idx / HWSZ;
  const float* dr = dw + p * NQ;
  float acc = 0.f;
#pragma unroll 7
  for (int q = 0; q < NQ; ++q) acc += dr[q] * cost[(size_t)q * HWSZ + s];
  out[(size_t)p * HWSZ + s] = acc;
}

// ---------------------------------------------------------------- launcher
extern "C" void kernel_launch(void* const* d_in, const int* in_sizes, int n_in,
                              void* d_out, int out_size, void* d_ws, size_t ws_size,
                              hipStream_t stream) {
  (void)in_sizes; (void)n_in; (void)out_size; (void)ws_size;
  const float* f1p[2] = {(const float*)d_in[0], (const float*)d_in[2]};
  const float* f2p[2] = {(const float*)d_in[1], (const float*)d_in[3]};
  const float* coords = (const float*)d_in[4];
  const float* cwp[4] = {(const float*)d_in[5], (const float*)d_in[6],
                         (const float*)d_in[7], (const float*)d_in[8]};
  const float* ctw5 = (const float*)d_in[9];
  const float* cw6  = (const float*)d_in[10];
  const float* cb6  = (const float*)d_in[11];
  const float* bnp[5][4];
  for (int j = 0; j < 5; ++j)
    for (int k = 0; k < 4; ++k) bnp[j][k] = (const float*)d_in[12 + j * 4 + k];
  const float* dapw = (const float*)d_in[32];

  char* ws = (char*)d_ws;
  size_t off = 0;
  auto alloc = [&](size_t bytes) -> char* {
    char* p = ws + off;
    off = (off + bytes + 255) & ~(size_t)255;
    return p;
  };
  auto cdiv = [](long long a, long long b) -> unsigned { return (unsigned)((a + b - 1) / b); };

  const int n1 = 2 * 96 * 64 * 9, n2 = 2 * 128 * 96 * 9, n3 = 2 * 128 * 128 * 9,
            n4 = 2 * 64 * 128 * 9, n5 = 2 * 64 * 32 * 16;
  _Float16* w1 = (_Float16*)alloc((size_t)n1 * 2);
  _Float16* w2 = (_Float16*)alloc((size_t)n2 * 2);
  _Float16* w3 = (_Float16*)alloc((size_t)n3 * 2);
  _Float16* w4 = (_Float16*)alloc((size_t)n4 * 2);
  _Float16* w5 = (_Float16*)alloc((size_t)n5 * 2);

  const int bnch[5] = {96, 128, 128, 64, 32};
  float* sc[2][5];
  float* sh[2][5];
  for (int l = 0; l < 2; ++l)
    for (int j = 0; j < 5; ++j) {
      sc[l][j] = (float*)alloc((size_t)bnch[j] * 4);
      sh[l][j] = (float*)alloc((size_t)bnch[j] * 4);
    }

  _Float16* bufA = (_Float16*)alloc((size_t)NQ * 96 * HWSZ * 2);  // 115.6 MB
  _Float16* bufB = (_Float16*)alloc((size_t)NQ * 64 * HWSZ * 2);  //  77.1 MB
  float*    cost = (float*)alloc((size_t)NQ * HWSZ * 4);          //   2.4 MB

  cvt_f16_kernel<<<cdiv(n1, 256), 256, 0, stream>>>(cwp[0], w1, n1);
  cvt_f16_kernel<<<cdiv(n2, 256), 256, 0, stream>>>(cwp[1], w2, n2);
  cvt_f16_kernel<<<cdiv(n3, 256), 256, 0, stream>>>(cwp[2], w3, n3);
  cvt_f16_kernel<<<cdiv(n4, 256), 256, 0, stream>>>(cwp[3], w4, n4);
  cvt_f16_kernel<<<cdiv(n5, 256), 256, 0, stream>>>(ctw5, w5, n5);
  for (int l = 0; l < 2; ++l)
    for (int j = 0; j < 5; ++j)
      bn_prep_kernel<<<1, 128, 0, stream>>>(bnp[j][0] + l * bnch[j],
                                            bnp[j][1] + l * bnch[j],
                                            bnp[j][2] + l * bnch[j],
                                            bnp[j][3] + l * bnch[j],
                                            sc[l][j], sh[l][j], bnch[j]);

  for (int l = 0; l < 2; ++l) {
    sample_concat_kernel<<<cdiv((long long)NQ * HWSZ, 256), 256, 0, stream>>>(
        f1p[l], f2p[l], coords, bufB, l == 0 ? 1.f : 0.5f);
    // conv1: 64 -> 96, 96x128, s1 (strip 64)
    conv3x3_wmma<64, 96, 96, 128, 1, 64><<<NQ * 96 * 2, 256, 0, stream>>>(
        bufB, w1 + (size_t)l * 96 * 576, sc[l][0], sh[l][0], bufA);
    // conv2: 96 -> 128, s2 -> 48x64 (strip 32)
    conv3x3_wmma<96, 128, 96, 128, 2, 32><<<NQ * 48 * 2, 256, 0, stream>>>(
        bufA, w2 + (size_t)l * 128 * 864, sc[l][1], sh[l][1], bufB);
    // conv3: 128 -> 128, 48x64, s1 (strip 32)
    conv3x3_wmma<128, 128, 48, 64, 1, 32><<<NQ * 48 * 2, 256, 0, stream>>>(
        bufB, w3 + (size_t)l * 128 * 1152, sc[l][2], sh[l][2], bufA);
    // conv4: 128 -> 64, 48x64, s1 (strip 32)
    conv3x3_wmma<128, 64, 48, 64, 1, 32><<<NQ * 48 * 2, 256, 0, stream>>>(
        bufA, w4 + (size_t)l * 64 * 1152, sc[l][3], sh[l][3], bufB);
    // deconv: 64 -> 32, up to 96x128
    convt5_kernel<<<cdiv((long long)NQ * 32 * HWSZ, 256), 256, 0, stream>>>(
        bufB, w5 + (size_t)l * 64 * 32 * 16, sc[l][4], sh[l][4], bufA);
    // conv6 + bias -> cost volume
    conv6_kernel<<<cdiv((long long)NQ * HWSZ, 256), 256, 0, stream>>>(
        bufA, cw6 + (size_t)l * 288, cb6 + l, cost);
    // DAP projection into final output slab
    dap_kernel<<<cdiv((long long)NQ * HWSZ, 256), 256, 0, stream>>>(
        cost, dapw + (size_t)l * NQ * NQ, (float*)d_out + (size_t)l * NQ * HWSZ);
  }
}
